// Rnn_81681688035946
// MI455X (gfx1250) — matrix operand order repaired and checked by reading, not compiled, not run on hardware
//
#include <hip/hip_runtime.h>
#include <hip/hip_bf16.h>
#include <math.h>

#define B_ 32
#define S_ 64
#define H_ 256
#define O_ 32000

typedef unsigned int   u32;
typedef unsigned short u16;
typedef __attribute__((ext_vector_type(4)))  u32    u32x4;
typedef __attribute__((ext_vector_type(16))) __bf16 v16bf;
typedef __attribute__((ext_vector_type(8)))  float  v8f;

union FragU { u32x4 q[2]; v16bf v; };

__device__ __forceinline__ u16 f32_to_bf16_rne(float f) {
    u32 u = __float_as_uint(f);
    u = u + 0x7FFFu + ((u >> 16) & 1u);   // round-to-nearest-even
    return (u16)(u >> 16);
}

// A-fragment (16x32 bf16, row-major source, ldk halfs per row).
// ISA layout: lanes 0-15 -> M=lane, K = {k0+0..7, k0+16..23}; lanes 16-31 -> K = {k0+8..15, k0+24..31}
__device__ __forceinline__ v16bf load_frag_A(const u16* base, int ldk, int m0, int k0, int lane) {
    int hi = lane >> 4;
    int m  = m0 + (lane & 15);
    const u16* p = base + (size_t)m * ldk + k0 + hi * 8;
    FragU f;
    f.q[0] = *(const u32x4*)(p);        // VGPR0-3: K = k0 + hi*8 + 0..7
    f.q[1] = *(const u32x4*)(p + 16);   // VGPR4-7: K = k0 + 16 + hi*8 + 0..7
    return f.v;
}

// B-fragment (32x16 bf16) from weight rows W[n][k] (row-major, ldk halfs).
// Lanes 0-15: N=lane, K={k0+0..7, k0+16..23} ; lanes 16-31: N=lane-16, K={k0+8..15, k0+24..31} (2 halfs per VGPR)
__device__ __forceinline__ v16bf load_frag_B(const u16* base, int ldk, int n0, int k0, int lane) {
    int hi = lane >> 4;
    int n  = n0 + (lane & 15);
    const u16* p = base + (size_t)n * ldk + k0 + hi * 8;
    FragU f;
    f.q[0] = *(const u32x4*)(p);
    f.q[1] = *(const u32x4*)(p + 16);
    return f.v;
}

// ---------------- elementwise prep kernels ----------------

__global__ void cvt_f32_bf16(const float* __restrict__ src, u16* __restrict__ dst, int n) {
    int i = blockIdx.x * blockDim.x + threadIdx.x;
    if (i < n) dst[i] = f32_to_bf16_rne(src[i]);
}

__global__ void make_bias(const float* __restrict__ a, const float* __restrict__ b,
                          float* __restrict__ o, int n) {
    int i = blockIdx.x * blockDim.x + threadIdx.x;
    if (i < n) o[i] = a[i] + b[i];
}

// xe_bf[(s*B + b)*H + k] = bf16(emb[x[b][s]][k])   (time-major rows, matches xp rows)
__global__ void gather_embed(const int* __restrict__ x, const float* __restrict__ emb,
                             u16* __restrict__ xe, int n) {
    int idx = blockIdx.x * blockDim.x + threadIdx.x;
    if (idx >= n) return;
    int k   = idx & (H_ - 1);
    int row = idx >> 8;          // s*B_ + b
    int b   = row & (B_ - 1);
    int s   = row >> 5;
    int tok = x[b * S_ + s];
    xe[idx] = f32_to_bf16_rne(emb[(size_t)tok * H_ + k]);
}

// ---------------- xp = xe @ W_ih^T + (b_ih+b_hh) :  [2048x256]x[256x256] ----------------

__global__ void __launch_bounds__(256) xp_gemm(const u16* __restrict__ xeb,
                                               const u16* __restrict__ wihb,
                                               const float* __restrict__ bb,
                                               float* __restrict__ xp) {
    int lane = threadIdx.x & 31;
    int gw   = blockIdx.x * (blockDim.x >> 5) + (threadIdx.x >> 5);  // 0..2047
    int mt = gw >> 4, nt = gw & 15;
    int m0 = mt * 16, n0 = nt * 16;
    v8f acc = {};
#pragma unroll
    for (int c = 0; c < 8; ++c) {
        v16bf a = load_frag_A(xeb,  H_, m0, c * 32, lane);
        v16bf b = load_frag_B(wihb, H_, n0, c * 32, lane);
        acc = __builtin_amdgcn_wmma_f32_16x16x32_bf16(false, a, false, b, (short)0, acc, false, false);
    }
    int hi = lane >> 4;
    int n  = n0 + (lane & 15);
    float bias = bb[n];
#pragma unroll
    for (int r = 0; r < 8; ++r) {
        int m = m0 + r + hi * 8;
        xp[(size_t)m * H_ + n] = acc[r] + bias;
    }
}

// ---------------- sequential recurrence: 4096 steps on ONE WGP ----------------
// 32 waves; wave w owns output tile (mt = w&1, nt = w>>1). W_hh B-fragments pinned in VGPRs.
// h double-buffered in LDS as bf16; xp loads software-pipelined under the per-step barrier.
// Captures sel[i] = g_{i,i} (bf16) and h_final (f32).

__global__ void __launch_bounds__(1024) rnn_recur(const u16* __restrict__ whhb,
                                                  const float* __restrict__ xp,
                                                  u16* __restrict__ sel_bf,
                                                  float* __restrict__ hfinal) {
    __shared__ u16 hbuf[2 * B_ * H_];   // 32 KB, double-buffered bf16 hidden state
    int lane = threadIdx.x & 31;
    int w    = threadIdx.x >> 5;        // 0..31
    int mt = w & 1, nt = w >> 1;
    int m0 = mt * 16, n0 = nt * 16;
    int hi = lane >> 4;
    int n  = n0 + (lane & 15);

    // Pin W_hh fragments for this wave's n-tile in registers for all 4096 steps.
    v16bf bw[8];
#pragma unroll
    for (int c = 0; c < 8; ++c) bw[c] = load_frag_B(whhb, H_, n0, c * 32, lane);

    for (int i = threadIdx.x; i < B_ * H_; i += blockDim.x) hbuf[i] = 0;  // h0 = 0

    // preload xp for t=0 (software pipeline stage 0)
    float xpv[8];
#pragma unroll
    for (int r = 0; r < 8; ++r) {
        int m = m0 + r + hi * 8;
        xpv[r] = xp[(size_t)m * H_ + n];
    }
    __syncthreads();

    int cur = 0;
    for (int i = 0; i < S_; ++i) {
        for (int t = 0; t < S_; ++t) {
            const u16* hc  = &hbuf[cur * B_ * H_];
            u16*       hnx = &hbuf[(cur ^ 1) * B_ * H_];
            // two independent accumulator chains -> halves WMMA RAW latency chain
            v8f acc0 = {}, acc1 = {};
#pragma unroll
            for (int c = 0; c < 4; ++c) {
                v16bf a = load_frag_A(hc, H_, m0, c * 32, lane);
                acc0 = __builtin_amdgcn_wmma_f32_16x16x32_bf16(false, a, false, bw[c], (short)0, acc0, false, false);
            }
#pragma unroll
            for (int c = 4; c < 8; ++c) {
                v16bf a = load_frag_A(hc, H_, m0, c * 32, lane);
                acc1 = __builtin_amdgcn_wmma_f32_16x16x32_bf16(false, a, false, bw[c], (short)0, acc1, false, false);
            }
            float hv[8];
#pragma unroll
            for (int r = 0; r < 8; ++r)
                hv[r] = tanhf(acc0[r] + acc1[r] + xpv[r]);
#pragma unroll
            for (int r = 0; r < 8; ++r) {
                int m = m0 + r + hi * 8;
                hnx[m * H_ + n] = f32_to_bf16_rne(hv[r]);
            }
            if (t == i) {   // capture g_{i,i} for the output head
#pragma unroll
                for (int r = 0; r < 8; ++r) {
                    int m = m0 + r + hi * 8;
                    sel_bf[(size_t)(i * B_ + m) * H_ + n] = f32_to_bf16_rne(hv[r]);
                }
                if (i == S_ - 1) {   // g_{63,63} == h_final
#pragma unroll
                    for (int r = 0; r < 8; ++r) {
                        int m = m0 + r + hi * 8;
                        hfinal[m * H_ + n] = hv[r];
                    }
                }
            }
            // prefetch next step's xp under the barrier + next WMMA chain
            int tn = (t == S_ - 1) ? 0 : t + 1;
#pragma unroll
            for (int r = 0; r < 8; ++r) {
                int m = m0 + r + hi * 8;
                xpv[r] = xp[(size_t)(tn * B_ + m) * H_ + n];
            }
            __syncthreads();
            cur ^= 1;
        }
    }
}

// ---------------- Y = sel @ W_out^T + b_out : [2048x256]x[256x32000] ----------------
// 4 M-tiles x 4 N-tiles per wave: B-fragments reused 4x -> L2 read traffic ~525 MB,
// balanced against the 262 MB non-temporal HBM store floor (~11.3 us @ 23.3 TB/s).

#define MTW 4   // M-tiles per wave
#define NTW 4   // N-tiles per wave

__global__ void __launch_bounds__(256) out_gemm(const u16* __restrict__ selb,
                                                const u16* __restrict__ woutb,
                                                const float* __restrict__ bout,
                                                float* __restrict__ ys) {
    int lane = threadIdx.x & 31;
    int gw   = blockIdx.x * (blockDim.x >> 5) + (threadIdx.x >> 5);  // 0..15999
    const int ngrp = (O_ / 16) / NTW;    // 500
    int mg = gw / ngrp;                  // 0..31
    int ng = gw % ngrp;
    int m0base = mg * (16 * MTW);        // 64 rows
    int n0base = ng * (16 * NTW);        // 64 cols

    v8f zero = {0.f, 0.f, 0.f, 0.f, 0.f, 0.f, 0.f, 0.f};
    v8f acc[MTW][NTW];
#pragma unroll
    for (int mi = 0; mi < MTW; ++mi)
#pragma unroll
        for (int j = 0; j < NTW; ++j) acc[mi][j] = zero;

#pragma unroll
    for (int c = 0; c < 8; ++c) {
        v16bf a[MTW], b[NTW];
#pragma unroll
        for (int mi = 0; mi < MTW; ++mi)
            a[mi] = load_frag_A(selb, H_, m0base + mi * 16, c * 32, lane);
#pragma unroll
        for (int j = 0; j < NTW; ++j)
            b[j] = load_frag_B(woutb, H_, n0base + j * 16, c * 32, lane);
#pragma unroll
        for (int mi = 0; mi < MTW; ++mi)
#pragma unroll
            for (int j = 0; j < NTW; ++j)
                acc[mi][j] = __builtin_amdgcn_wmma_f32_16x16x32_bf16(
                    false, a[mi], false, b[j], (short)0, acc[mi][j], false, false);
    }

    int hi = lane >> 4;
#pragma unroll
    for (int j = 0; j < NTW; ++j) {
        int n = n0base + j * 16 + (lane & 15);
        float bo = bout[n];
#pragma unroll
        for (int mi = 0; mi < MTW; ++mi) {
#pragma unroll
            for (int r = 0; r < 8; ++r) {
                int m  = m0base + mi * 16 + r + hi * 8;   // row = i*B + b
                int ii = m >> 5;
                int bb = m & 31;
                // stream to HBM, don't pollute L2 (keep W_out resident)
                __builtin_nontemporal_store(acc[mi][j][r] + bo,
                                            &ys[((size_t)bb * S_ + ii) * O_ + n]);
            }
        }
    }
}

// ---------------- launch ----------------

extern "C" void kernel_launch(void* const* d_in, const int* in_sizes, int n_in,
                              void* d_out, int out_size, void* d_ws, size_t ws_size,
                              hipStream_t stream) {
    (void)in_sizes; (void)n_in; (void)out_size; (void)ws_size;
    const int*   x    = (const int*)d_in[0];
    const float* emb  = (const float*)d_in[1];
    const float* Wih  = (const float*)d_in[2];
    const float* Whh  = (const float*)d_in[3];
    const float* bih  = (const float*)d_in[4];
    const float* bhh  = (const float*)d_in[5];
    const float* Wout = (const float*)d_in[6];
    const float* bout = (const float*)d_in[7];

    float* out    = (float*)d_out;
    float* hfinal = out;               // [1,B,H] = 8192 floats
    float* ys     = out + B_ * H_;     // [B,S,O]

    char* ws = (char*)d_ws;            // all offsets 256B-aligned
    u16*   wih_b  = (u16*)  (ws);                  //   131,072 B
    u16*   whh_b  = (u16*)  (ws +   131072);       //   131,072 B
    u16*   wout_b = (u16*)  (ws +   262144);       // 16,384,000 B
    float* bb     = (float*)(ws + 16646144);       //     1,024 B
    u16*   xe_b   = (u16*)  (ws + 16647168);       // 1,048,576 B
    float* xp     = (float*)(ws + 17695744);       // 2,097,152 B
    u16*   sel_b  = (u16*)  (ws + 19792896);       // 1,048,576 B  (total ~20.8 MB)

    cvt_f32_bf16<<<(H_ * H_ + 255) / 256, 256, 0, stream>>>(Wih, wih_b, H_ * H_);
    cvt_f32_bf16<<<(H_ * H_ + 255) / 256, 256, 0, stream>>>(Whh, whh_b, H_ * H_);
    cvt_f32_bf16<<<(O_ * H_ + 255) / 256, 256, 0, stream>>>(Wout, wout_b, O_ * H_);
    make_bias<<<1, 256, 0, stream>>>(bih, bhh, bb, H_);
    gather_embed<<<(S_ * B_ * H_) / 256, 256, 0, stream>>>(x, emb, xe_b, S_ * B_ * H_);

    // xp GEMM: 128 M-tiles x 16 N-tiles = 2048 wave-tiles / 8 waves per block
    xp_gemm<<<256, 256, 0, stream>>>(xe_b, wih_b, bb, xp);

    // sequential recurrence: one workgroup, 32 waves, one WGP
    rnn_recur<<<1, 1024, 0, stream>>>(whh_b, xp, sel_b, hfinal);

    // output projection: 32 M-groups x 500 N-groups = 16000 waves / 8 per block
    out_gemm<<<2000, 256, 0, stream>>>(sel_b, wout_b, bout, ys);
}
